// ActorCritic_19104014532995
// MI455X (gfx1250) — matrix operand order repaired and checked by reading, not compiled, not run on hardware
//
#include <hip/hip_runtime.h>
#include <hip/hip_bf16.h>

// ---------------- problem constants (match reference) ----------------
#define NNODES   131072      // N = B*(L+1)
#define NEDGES   524288      // E (random edges); +NNODES self loops appended
#define NBATCH   2048        // B
#define CDIM     64          // per-head channels (all layers)
#define ETOT     (NEDGES + NNODES)

// ---------------- types for WMMA ----------------
typedef __attribute__((ext_vector_type(16))) __bf16 v16bf;
typedef __attribute__((ext_vector_type(8)))  float  v8f;
typedef __attribute__((ext_vector_type(8)))  unsigned vu8;

__device__ __forceinline__ unsigned short bfbits(float f) {
  unsigned u = __builtin_bit_cast(unsigned, f);
  unsigned r = u + 0x7FFFu + ((u >> 16) & 1u);    // round-to-nearest-even
  return (unsigned short)(r >> 16);
}

// monotone float <-> uint encoding for atomicMax-based segment max
__device__ __forceinline__ unsigned enc_f(float f) {
  unsigned b = __builtin_bit_cast(unsigned, f);
  return (b & 0x80000000u) ? ~b : (b | 0x80000000u);
}
__device__ __forceinline__ float dec_f(unsigned u) {
  unsigned b = (u & 0x80000000u) ? (u & 0x7FFFFFFFu) : ~u;
  return __builtin_bit_cast(float, b);
}

// ---------------- fp32 -> packed bf16 (2 per dword) ----------------
__global__ void cvt_bf16(const float* __restrict__ in, unsigned* __restrict__ out,
                         long long npairs) {
  long long i = (long long)blockIdx.x * blockDim.x + threadIdx.x;
  if (i >= npairs) return;
  unsigned lo = bfbits(in[2 * i + 0]);
  unsigned hi = bfbits(in[2 * i + 1]);
  out[i] = lo | (hi << 16);
}

// ---------------- W [K,Nc] fp32 -> Wt [Nc,K] bf16 ----------------
__global__ void wt_bf16(const float* __restrict__ W, unsigned short* __restrict__ Wt,
                        int K, int Nc) {
  int idx = blockIdx.x * blockDim.x + threadIdx.x;
  if (idx >= K * Nc) return;
  int k = idx / Nc, n = idx - k * Nc;             // coalesced read of W
  Wt[(size_t)n * K + k] = bfbits(W[idx]);
}

// =====================================================================
// GEMM: C[M,Nc] = A[M,K] @ W[K,Nc]; A is bf16 [M,K], Bt is bf16 [Nc,K].
// One wave -> 16x64 output strip (4 WMMAs / K-step, A fragment reused).
// A frag (ISA 16-bit A 16x32): lane=half*16+m; dwords [k0/2+half*4 ..+3],
//   [k0/2+8+half*4 ..+3]  -> two b128 loads.
// B frag (32x16, same K order as A along Wt row) -> dwords at
//   ncol*K/2 + k0/2 + half*4 ..+3 and k0/2 + 8 + half*4 ..+3 -> two b128 loads.
// C/D: VGPR v -> row (v + half*8), col = lane&15.
// =====================================================================
__global__ __launch_bounds__(256)
void wmma_gemm_bf16(const unsigned short* __restrict__ Ab,
                    const unsigned short* __restrict__ Bt,
                    float* __restrict__ Cc, int M, int K, int Nc) {
  const int wave   = (blockIdx.x * (blockDim.x >> 5)) + (threadIdx.x >> 5);
  const int tilesN = (Nc + 63) >> 6;
  const int total  = (M >> 4) * tilesN;
  if (wave >= total) return;
  const int tm   = wave / tilesN;
  const int tg   = wave % tilesN;
  const int lane = threadIdx.x & 31;
  const int half = lane >> 4;
  const int mr   = lane & 15;

  const unsigned* Arow = (const unsigned*)(Ab + (size_t)(tm * 16 + mr) * K);
  v8f acc[4] = {};

  for (int k0 = 0; k0 < K; k0 += 32) {
    const int kd = k0 >> 1;
    const uint4 alo = *(const uint4*)(Arow + kd + half * 4);
    const uint4 ahi = *(const uint4*)(Arow + kd + 8 + half * 4);
    vu8 av = {alo.x, alo.y, alo.z, alo.w, ahi.x, ahi.y, ahi.z, ahi.w};
    v16bf a = __builtin_bit_cast(v16bf, av);
#pragma unroll
    for (int t = 0; t < 4; ++t) {
      const int ncol = tg * 64 + t * 16 + mr;
      if (tg * 64 + t * 16 < Nc) {                 // wave-uniform guard
        const unsigned* Brow = (const unsigned*)(Bt + (size_t)ncol * K);
        const uint4 blo = *(const uint4*)(Brow + kd + half * 4);
        const uint4 bhi = *(const uint4*)(Brow + kd + 8 + half * 4);
        vu8 bv = {blo.x, blo.y, blo.z, blo.w, bhi.x, bhi.y, bhi.z, bhi.w};
        v16bf b = __builtin_bit_cast(v16bf, bv);
        acc[t] = __builtin_amdgcn_wmma_f32_16x16x32_bf16(false, a, false, b,
                                                         (short)0, acc[t],
                                                         false, false);
      }
    }
  }
#pragma unroll
  for (int t = 0; t < 4; ++t) {
    if (tg * 64 + t * 16 < Nc) {
      const int col = tg * 64 + t * 16 + mr;
#pragma unroll
      for (int v = 0; v < 8; ++v) {
        const int row = tm * 16 + v + half * 8;
        Cc[(size_t)row * Nc + col] = acc[t][v];
      }
    }
  }
}

// ---------------- per-node attention coefficients ----------------
__global__ void attn_coeff(const float* __restrict__ h,
                           const float* __restrict__ a_s,
                           const float* __restrict__ a_d,
                           float* __restrict__ als, float* __restrict__ ald,
                           int n, int H) {
  int idx = blockIdx.x * blockDim.x + threadIdx.x;
  if (idx >= n * H) return;
  int node = idx / H, hh = idx - node * H;
  const float* hp = h + (size_t)node * H * CDIM + hh * CDIM;
  const float* ps = a_s + hh * CDIM;
  const float* pd = a_d + hh * CDIM;
  float ss = 0.f, sd = 0.f;
#pragma unroll 8
  for (int c = 0; c < CDIM; ++c) { float v = hp[c]; ss += v * ps[c]; sd += v * pd[c]; }
  als[idx] = ss; ald[idx] = sd;
}

// ---------------- fill ----------------
__global__ void fill_u32(unsigned* __restrict__ p, unsigned v, long long n) {
  long long i = (long long)blockIdx.x * blockDim.x + threadIdx.x;
  if (i < n) p[i] = v;
}

// ---------------- edge endpoints (self-loops appended after E edges) ----
__device__ __forceinline__ void edge_sd(const int* __restrict__ ei, long long e,
                                        int& s, int& d) {
  if (e < NEDGES) { s = ei[e]; d = ei[NEDGES + e]; }
  else            { s = (int)(e - NEDGES); d = s; }
}

// pass 1: segment max of leaky-relu logits (encoded-uint atomicMax)
__global__ void edge_max_k(const int* __restrict__ ei,
                           const float* __restrict__ als, const float* __restrict__ ald,
                           unsigned* __restrict__ menc, int H) {
  long long idx = (long long)blockIdx.x * blockDim.x + threadIdx.x;
  if (idx >= (long long)ETOT * H) return;
  long long e = idx / H; int hh = (int)(idx - e * H);
  int s, d; edge_sd(ei, e, s, d);
  float v = als[s * H + hh] + ald[d * H + hh];
  v = v > 0.f ? v : 0.2f * v;
  atomicMax(&menc[d * H + hh], enc_f(v));
}

// pass 2: segment sum of exp(e - max)
__global__ void edge_sum_k(const int* __restrict__ ei,
                           const float* __restrict__ als, const float* __restrict__ ald,
                           const unsigned* __restrict__ menc, float* __restrict__ ssum,
                           int H) {
  long long idx = (long long)blockIdx.x * blockDim.x + threadIdx.x;
  if (idx >= (long long)ETOT * H) return;
  long long e = idx / H; int hh = (int)(idx - e * H);
  int s, d; edge_sd(ei, e, s, d);
  float v = als[s * H + hh] + ald[d * H + hh];
  v = v > 0.f ? v : 0.2f * v;
  float w = __expf(v - dec_f(menc[d * H + hh]));
  atomicAdd(&ssum[d * H + hh], w);
}

// pass 3: weighted scatter  agg[dst,f] += h[src,f] * alpha(e,h)
__global__ void edge_scatter_k(const int* __restrict__ ei,
                               const float* __restrict__ als, const float* __restrict__ ald,
                               const unsigned* __restrict__ menc, const float* __restrict__ ssum,
                               const float* __restrict__ h, float* __restrict__ agg, int H) {
  const int HC = H * CDIM;
  long long idx = (long long)blockIdx.x * blockDim.x + threadIdx.x;
  if (idx >= (long long)ETOT * HC) return;
  long long e = idx / HC; int f = (int)(idx - e * HC); int hh = f >> 6;
  int s, d; edge_sd(ei, e, s, d);
  float v = als[s * H + hh] + ald[d * H + hh];
  v = v > 0.f ? v : 0.2f * v;
  float w = __expf(v - dec_f(menc[d * H + hh]));
  float alpha = w / (ssum[d * H + hh] + 1e-16f);
  atomicAdd(&agg[(size_t)d * HC + f], h[(size_t)s * HC + f] * alpha);
}

// ---------------- bias + relu (in place) ----------------
__global__ void bias_relu(float* __restrict__ x, const float* __restrict__ b,
                          long long n, int F) {
  long long i = (long long)blockIdx.x * blockDim.x + threadIdx.x;
  if (i >= n * F) return;
  float v = x[i] + b[(int)(i % F)];
  x[i] = v > 0.f ? v : 0.f;
}

// ---------------- fc3 + softmax (one thread per batch row) ----------------
__global__ void fc3_softmax(const float* __restrict__ hm2, const float* __restrict__ W,
                            const float* __restrict__ b, float* __restrict__ out, int Bn) {
  int row = blockIdx.x * blockDim.x + threadIdx.x;
  if (row >= Bn) return;
  const float* hp = hm2 + row * 32;
  float o[16]; float mx = -1e30f;
#pragma unroll
  for (int j = 0; j < 16; ++j) {
    float acc = b[j];
#pragma unroll
    for (int k = 0; k < 32; ++k) acc += hp[k] * W[k * 16 + j];
    o[j] = acc; mx = fmaxf(mx, acc);
  }
  float ssum = 0.f;
#pragma unroll
  for (int j = 0; j < 16; ++j) { o[j] = __expf(o[j] - mx); ssum += o[j]; }
  float inv = 1.f / ssum;
#pragma unroll
  for (int j = 0; j < 16; ++j) out[row * 16 + j] = o[j] * inv;
}

// =====================================================================
static inline int cdiv(long long a, long long b) { return (int)((a + b - 1) / b); }

extern "C" void kernel_launch(void* const* d_in, const int* in_sizes, int n_in,
                              void* d_out, int out_size, void* d_ws, size_t ws_size,
                              hipStream_t stream) {
  const float* x0   = (const float*)d_in[0];
  const int*   ei   = (const int*)  d_in[1];
  const float* W1   = (const float*)d_in[2];
  const float* a1s  = (const float*)d_in[3];
  const float* a1d  = (const float*)d_in[4];
  const float* b1   = (const float*)d_in[5];
  const float* W2   = (const float*)d_in[6];
  const float* a2s  = (const float*)d_in[7];
  const float* a2d  = (const float*)d_in[8];
  const float* b2   = (const float*)d_in[9];
  const float* W3   = (const float*)d_in[10];
  const float* a3s  = (const float*)d_in[11];
  const float* a3d  = (const float*)d_in[12];
  const float* b3   = (const float*)d_in[13];
  const float* fcW1 = (const float*)d_in[14];
  const float* fcb1 = (const float*)d_in[15];
  const float* fcW2 = (const float*)d_in[16];
  const float* fcb2 = (const float*)d_in[17];
  const float* fcW3 = (const float*)d_in[18];
  const float* fcb3 = (const float*)d_in[19];
  float* out = (float*)d_out;

  // ---------------- workspace carve (16B-aligned chunks) ----------------
  char* w = (char*)d_ws;
  float*          hbuf = (float*)w;          w += (size_t)NNODES * 256 * 4;
  float*          xbuf = (float*)w;          w += (size_t)NNODES * 256 * 4;
  unsigned short* xb   = (unsigned short*)w; w += (size_t)NNODES * 256 * 2;  // bf16 activations
  unsigned short* wtb  = (unsigned short*)w; w += (size_t)4096 * 64 * 2;     // bf16 Wt (max fcW1)
  float*          als  = (float*)w;          w += (size_t)NNODES * 4 * 4;
  float*          ald  = (float*)w;          w += (size_t)NNODES * 4 * 4;
  unsigned*       menc = (unsigned*)w;       w += (size_t)NNODES * 4 * 4;
  float*          ssum = (float*)w;          w += (size_t)NNODES * 4 * 4;
  float*          hm1  = (float*)w;          w += (size_t)NBATCH * 64 * 4;
  float*          hm2  = (float*)w;          w += (size_t)NBATCH * 32 * 4;

  const int T = 256;

  // cvt A -> bf16 (xb), transpose W -> bf16 (wtb), then WMMA GEMM
  auto gemm = [&](const float* A, const float* W, float* C, int M, int K, int Nc) {
    long long npairs = (long long)M * K / 2;
    cvt_bf16<<<cdiv(npairs, T), T, 0, stream>>>(A, (unsigned*)xb, npairs);
    wt_bf16<<<cdiv((long long)K * Nc, T), T, 0, stream>>>(W, wtb, K, Nc);
    int tiles = (M / 16) * ((Nc + 63) / 64);
    wmma_gemm_bf16<<<cdiv(tiles, 8), T, 0, stream>>>(xb, wtb, C, M, K, Nc);
  };

  // one GAT layer's edge phase: h[N,H*64] -> agg = relu(softmax-agg + bias)
  auto gat_edges = [&](const float* h, const float* as_, const float* ad_,
                       const float* bias, int H, float* agg) {
    const int HC = H * CDIM;
    attn_coeff<<<cdiv((long long)NNODES * H, T), T, 0, stream>>>(h, as_, ad_, als, ald, NNODES, H);
    fill_u32<<<cdiv((long long)NNODES * H, T), T, 0, stream>>>(menc, 0u, (long long)NNODES * H);
    fill_u32<<<cdiv((long long)NNODES * H, T), T, 0, stream>>>((unsigned*)ssum, 0u, (long long)NNODES * H);
    fill_u32<<<cdiv((long long)NNODES * HC, T), T, 0, stream>>>((unsigned*)agg, 0u, (long long)NNODES * HC);
    edge_max_k<<<cdiv((long long)ETOT * H, T), T, 0, stream>>>(ei, als, ald, menc, H);
    edge_sum_k<<<cdiv((long long)ETOT * H, T), T, 0, stream>>>(ei, als, ald, menc, ssum, H);
    edge_scatter_k<<<cdiv((long long)ETOT * HC, T), T, 0, stream>>>(ei, als, ald, menc, ssum, h, agg, H);
    bias_relu<<<cdiv((long long)NNODES * HC, T), T, 0, stream>>>(agg, bias, NNODES, HC);
  };

  // ---- GAT layer 1: F=32 -> 4 heads x 64 (concat 256) ----
  gemm(x0, W1, hbuf, NNODES, 32, 256);
  gat_edges(hbuf, a1s, a1d, b1, 4, xbuf);

  // ---- GAT layer 2: 256 -> 64, 1 head ----
  gemm(xbuf, W2, hbuf, NNODES, 256, 64);
  gat_edges(hbuf, a2s, a2d, b2, 1, xbuf);

  // ---- GAT layer 3: 64 -> 64, 1 head ----
  gemm(xbuf, W3, hbuf, NNODES, 64, 64);
  gat_edges(hbuf, a3s, a3d, b3, 1, xbuf);

  // ---- actor MLP: x_gat [2048, 4096] (row-major view of xbuf[N,64]) ----
  gemm(xbuf, fcW1, hm1, NBATCH, 4096, 64);
  bias_relu<<<cdiv((long long)NBATCH * 64, T), T, 0, stream>>>(hm1, fcb1, NBATCH, 64);

  gemm(hm1, fcW2, hm2, NBATCH, 64, 32);
  bias_relu<<<cdiv((long long)NBATCH * 32, T), T, 0, stream>>>(hm2, fcb2, NBATCH, 32);

  fc3_softmax<<<cdiv(NBATCH, T), T, 0, stream>>>(hm2, fcW3, fcb3, out, NBATCH);
}
